// AWARE_33887291966079
// MI455X (gfx1250) — matrix operand order repaired and checked by reading, not compiled, not run on hardware
//
#include <hip/hip_runtime.h>
#include <hip/hip_bf16.h>
#include <cstdint>

// ---------------- problem constants ----------------
#define BB   64
#define NN   512
#define FEAT 256
#define EMB  128
#define RP   64
#define TT   8      // MAX_WALK_LEN
#define D0   512    // RP*TT

typedef __attribute__((ext_vector_type(16))) __bf16 v16bf;
typedef __attribute__((ext_vector_type(8)))  float  v8f;

__device__ __forceinline__ float fast_rcp(float x) {
#if defined(__gfx1250__)
  return __builtin_amdgcn_rcpf(x);  // single v_rcp_f32, ~1 ulp
#else
  return 1.0f / x;
#endif
}
__device__ __forceinline__ float sigmoidf(float x) {
  return fast_rcp(1.0f + __expf(-x));
}

// ---- gfx1250 async global->LDS staging (guarded; falls back to plain copy) ----
#if defined(__gfx1250__) && __has_builtin(__builtin_amdgcn_global_load_async_to_lds_b128)
#define HAVE_ASYNC_LDS 1
#else
#define HAVE_ASYNC_LDS 0
#endif

typedef int v4i_async __attribute__((vector_size(16)));

__device__ __forceinline__ void stage16(__bf16* lds_dst, const __bf16* g_src) {
#if HAVE_ASYNC_LDS
  __builtin_amdgcn_global_load_async_to_lds_b128(
      (__attribute__((address_space(1))) v4i_async*)(void*)(const_cast<__bf16*>(g_src)),
      (__attribute__((address_space(3))) v4i_async*)(void*)lds_dst, 0, 0);
#else
  *(uint4*)lds_dst = *(const uint4*)g_src;
#endif
}
__device__ __forceinline__ void stage_commit() {
#if HAVE_ASYNC_LDS
  asm volatile("s_wait_asynccnt 0" ::: "memory");
#endif
  __syncthreads();
}

// ---- WMMA operand packing (ISA 16x16x32 bf16 layouts) ----
// A 16x32: lane row = lane&15; elem i<8 -> k=(lane>>4)*8+i ; i>=8 -> k=16+(lane>>4)*8+(i-8)
template <typename L>
__device__ __forceinline__ v16bf load_a16(int kbase, L f) {
  const int lane = threadIdx.x & 31, lo = lane & 15, hi = lane >> 4;
  v16bf a;
#pragma unroll
  for (int i = 0; i < 16; ++i) {
    const int ka = (i < 8) ? (hi * 8 + i) : (8 + hi * 8 + i);
    a[i] = f(lo, kbase + ka);
  }
  return a;
}
// B 32x16: lane col = lane&15; elem i<8 -> k=(lane>>4)*8+i ; i>=8 -> k=16+(lane>>4)*8+(i-8)
template <typename L>
__device__ __forceinline__ v16bf load_b16(int kbase, L f) {
  const int lane = threadIdx.x & 31, lo = lane & 15, hi = lane >> 4;
  v16bf b;
#pragma unroll
  for (int i = 0; i < 16; ++i) b[i] = f(kbase + ((i < 8) ? (hi * 8 + i) : (8 + hi * 8 + i)), lo);
  return b;
}
__device__ __forceinline__ v8f wmma_mma(const v16bf& a, const v16bf& b, const v8f& c) {
  return __builtin_amdgcn_wmma_f32_16x16x32_bf16(false, a, false, b, (short)0, c,
                                                 false, false);
}
// C/D 16x16 f32: lane col = lane&15, VGPR v -> row = v + (lane>>4)*8
template <typename ST>
__device__ __forceinline__ void wmma_store(const v8f& c, ST store) {
  const int lane = threadIdx.x & 31, lo = lane & 15, hi = lane >> 4;
#pragma unroll
  for (int v = 0; v < 8; ++v) store(hi * 8 + v, lo, c[v]);
}

// ---------------- kernels ----------------

// f32 -> bf16 conversion (weights)
__global__ void k_cvt(const float* __restrict__ src, __bf16* __restrict__ dst, int n) {
  int i = blockIdx.x * blockDim.x + threadIdx.x;
  if (i < n) dst[i] = (__bf16)src[i];
}

// Wc = Wv @ W (fold two front GEMMs), convert Ww/Wg, zero gate accumulator.
__global__ void k_prep(const float* __restrict__ W, const float* __restrict__ Wv,
                       const float* __restrict__ Ww, const float* __restrict__ Wg,
                       __bf16* __restrict__ wc, __bf16* __restrict__ wwb,
                       __bf16* __restrict__ wgb, float* __restrict__ fs) {
  const int t = threadIdx.x;  // one block, 256 threads
  for (int idx = t; idx < RP * FEAT; idx += 256) {
    const int r = idx / FEAT, f = idx % FEAT;
    float acc = 0.f;
    for (int e = 0; e < EMB; ++e) acc += Wv[r * EMB + e] * W[e * FEAT + f];
    wc[idx] = (__bf16)acc;
  }
  for (int idx = t; idx < RP * RP; idx += 256) {
    wwb[idx] = (__bf16)Ww[idx];
    wgb[idx] = (__bf16)Wg[idx];
  }
  for (int idx = t; idx < BB * TT * RP; idx += 256) fs[idx] = 0.f;
}

// F1[b] = sigmoid(Wc @ x[b]^T) : M=RP, N=NN, K=FEAT. grid(8, BB), block 128.
// x tile staged to LDS (f32->bf16 during copy); shared by all 4 waves.
__global__ void __launch_bounds__(128)
k_f1(const float* __restrict__ x, const __bf16* __restrict__ wc,
     float* __restrict__ F1f, __bf16* __restrict__ Fn0, __bf16* __restrict__ Fn0T) {
  __shared__ __align__(16) __bf16 xs[64 * FEAT];  // 32 KB: x[n0..n0+63][0..255] bf16
  const int b = blockIdx.y;
  const int n0 = blockIdx.x * 64;
  const int w = threadIdx.x >> 5;  // 4 waves -> row strip w*16
  const int r0 = w * 16;
  const float* xb = x + (size_t)b * NN * FEAT;
  for (int idx = threadIdx.x; idx < 64 * (FEAT / 4); idx += 128) {
    const int row = idx / (FEAT / 4);
    const int kc = (idx % (FEAT / 4)) * 4;
    const float4 v = *(const float4*)&xb[(size_t)(n0 + row) * FEAT + kc];
    xs[row * FEAT + kc + 0] = (__bf16)v.x;
    xs[row * FEAT + kc + 1] = (__bf16)v.y;
    xs[row * FEAT + kc + 2] = (__bf16)v.z;
    xs[row * FEAT + kc + 3] = (__bf16)v.w;
  }
  __syncthreads();

  v8f acc[4] = {};
  for (int k0 = 0; k0 < FEAT; k0 += 32) {
    const v16bf a = load_a16(k0, [&](int r, int k) { return wc[(r0 + r) * FEAT + k]; });
#pragma unroll
    for (int j = 0; j < 4; ++j) {
      const v16bf bb = load_b16(k0,
          [&](int k, int c) { return xs[(j * 16 + c) * FEAT + k]; });
      acc[j] = wmma_mma(a, bb, acc[j]);
    }
  }
  float* F1b_ = F1f + (size_t)b * RP * NN;
  __bf16* fn = Fn0 + (size_t)b * RP * NN;
  __bf16* fnT = Fn0T + (size_t)b * NN * RP;
#pragma unroll
  for (int j = 0; j < 4; ++j) {
    const int c0 = n0 + j * 16;
    wmma_store(acc[j], [&](int r, int c, float v) {
      const float s = sigmoidf(v);
      const int rr = r0 + r, cc = c0 + c;
      F1b_[(size_t)rr * NN + cc] = s;
      const __bf16 sb = (__bf16)s;
      fn[(size_t)rr * NN + cc] = sb;
      fnT[(size_t)cc * RP + rr] = sb;  // per-lane contiguous 8 rows
    });
  }
}

// fs[b,it,:] += sum_n sigmoid(Wg @ Fn)[:,n]. Wg staged in LDS. grid(4, BB), block 256.
__global__ void __launch_bounds__(256)
k_gate(const __bf16* __restrict__ FnT, const __bf16* __restrict__ wg,
       float* __restrict__ fs, int it) {
  __shared__ __align__(16) __bf16 wsh[RP * RP];  // 8 KB
  __shared__ float g[RP];
  const int b = blockIdx.y;
  const int m0 = blockIdx.x * 128;
  const int w = threadIdx.x >> 5;
  if (threadIdx.x < RP) g[threadIdx.x] = 0.f;
  for (int idx = threadIdx.x; idx < RP * RP / 8; idx += 256)
    ((uint4*)wsh)[idx] = ((const uint4*)wg)[idx];
  __syncthreads();
  const __bf16* fnT = FnT + (size_t)b * NN * RP;
  const int c0 = m0 + w * 16;
  v8f acc[4] = {};
  for (int k0 = 0; k0 < RP; k0 += 32) {
    const v16bf bb = load_b16(k0,
        [&](int k, int c) { return fnT[(size_t)(c0 + c) * RP + k]; });
#pragma unroll
    for (int j = 0; j < 4; ++j) {
      const v16bf a = load_a16(k0, [&](int r, int k) { return wsh[(j * 16 + r) * RP + k]; });
      acc[j] = wmma_mma(a, bb, acc[j]);
    }
  }
#pragma unroll
  for (int j = 0; j < 4; ++j)
    wmma_store(acc[j], [&](int r, int c, float v) {
      atomicAdd(&g[j * 16 + r], sigmoidf(v));
    });
  __syncthreads();
  if (threadIdx.x < RP)
    atomicAdd(&fs[(size_t)b * TT * RP + it * RP + threadIdx.x], g[threadIdx.x]);
}

// WwFT[m][r] = (Ww @ Fn)[r][m] transposed. Ww staged in LDS. grid(4, BB), block 256.
__global__ void __launch_bounds__(256)
k_wwf(const __bf16* __restrict__ FnT, const __bf16* __restrict__ ww,
      __bf16* __restrict__ WwFT) {
  __shared__ __align__(16) __bf16 wsh[RP * RP];  // 8 KB
  const int b = blockIdx.y;
  const int m0 = blockIdx.x * 128;
  const int w = threadIdx.x >> 5;
  for (int idx = threadIdx.x; idx < RP * RP / 8; idx += 256)
    ((uint4*)wsh)[idx] = ((const uint4*)ww)[idx];
  __syncthreads();
  const __bf16* fnT = FnT + (size_t)b * NN * RP;
  __bf16* outT = WwFT + (size_t)b * NN * RP;
  const int c0 = m0 + w * 16;
  v8f acc[4] = {};
  for (int k0 = 0; k0 < RP; k0 += 32) {
    const v16bf bb = load_b16(k0,
        [&](int k, int c) { return fnT[(size_t)(c0 + c) * RP + k]; });
#pragma unroll
    for (int j = 0; j < 4; ++j) {
      const v16bf a = load_a16(k0, [&](int r, int k) { return wsh[(j * 16 + r) * RP + k]; });
      acc[j] = wmma_mma(a, bb, acc[j]);
    }
  }
#pragma unroll
  for (int j = 0; j < 4; ++j)
    wmma_store(acc[j], [&](int r, int c, float v) {
      outT[(size_t)(c0 + c) * RP + (j * 16 + r)] = (__bf16)v;  // contiguous
    });
}

// ST[m][n] = adj[n][m] ? (Fn^T @ WwF)[n][m] : -1e8. WwFT tile async-staged to LDS.
// grid(16, BB), block 256.
__global__ void __launch_bounds__(256)
k_s(const __bf16* __restrict__ FnT, const __bf16* __restrict__ WwFT,
    const int* __restrict__ adj, float* __restrict__ ST) {
  __shared__ __align__(16) __bf16 bsh[128 * RP];  // 16 KB: WwFT[m0..m0+127][:]
  const int b = blockIdx.y;
  const int n0 = (blockIdx.x & 3) * 128;
  const int m0 = (blockIdx.x >> 2) * 128;
  const int w = threadIdx.x >> 5;
  const __bf16* fnT = FnT + (size_t)b * NN * RP;
  const __bf16* wfT = WwFT + (size_t)b * NN * RP;
  for (int idx = threadIdx.x; idx < 128 * RP / 8; idx += 256)
    stage16(&bsh[idx * 8], &wfT[(size_t)m0 * RP + idx * 8]);
  stage_commit();

  const int* ad = adj + (size_t)b * NN * NN;
  float* Sb = ST + (size_t)b * NN * NN;
  const int r0 = n0 + w * 16;
  const v16bf a0 = load_a16(0, [&](int r, int k) { return fnT[(size_t)(r0 + r) * RP + k]; });
  const v16bf a1 = load_a16(32, [&](int r, int k) { return fnT[(size_t)(r0 + r) * RP + k]; });
#pragma unroll
  for (int j = 0; j < 8; ++j) {
    const int c0 = m0 + j * 16;
    const v16bf b0 = load_b16(0,
        [&](int k, int c) { return bsh[(j * 16 + c) * RP + k]; });
    const v16bf b1 = load_b16(32,
        [&](int k, int c) { return bsh[(j * 16 + c) * RP + k]; });
    v8f acc = {};
    acc = wmma_mma(a0, b0, acc);
    acc = wmma_mma(a1, b1, acc);
    wmma_store(acc, [&](int r, int c, float v) {
      const int n = r0 + r, m = c0 + c;
      Sb[(size_t)m * NN + n] = ad[(size_t)n * NN + m] ? v : -1e8f;  // row write
    });
  }
}

// AT[m][n] = softmax_n(ST[m][n]); one wave per m, fully coalesced. grid(64, BB), block 256.
__global__ void __launch_bounds__(256)
k_softmax(const float* __restrict__ ST, __bf16* __restrict__ AT) {
  const int b = blockIdx.y;
  const int m = blockIdx.x * 8 + (threadIdx.x >> 5);
  const int lane = threadIdx.x & 31;
  const float* row = ST + (size_t)b * NN * NN + (size_t)m * NN;
  float vals[16];
  float mx = -3.4e38f;
#pragma unroll
  for (int i = 0; i < 16; ++i) {
    vals[i] = row[lane + i * 32];
    mx = fmaxf(mx, vals[i]);
  }
#pragma unroll
  for (int off = 16; off; off >>= 1) mx = fmaxf(mx, __shfl_xor(mx, off));
  float sum = 0.f;
#pragma unroll
  for (int i = 0; i < 16; ++i) {
    vals[i] = __expf(vals[i] - mx);
    sum += vals[i];
  }
#pragma unroll
  for (int off = 16; off; off >>= 1) sum += __shfl_xor(sum, off);
  const float inv = fast_rcp(sum);
  __bf16* arow = AT + (size_t)b * NN * NN + (size_t)m * NN;
#pragma unroll
  for (int i = 0; i < 16; ++i) arow[lane + i * 32] = (__bf16)(vals[i] * inv);
}

// Fn_next = (Fn @ A) * F1 : M=RP, N=NN, K=NN. Full Fn async-staged to LDS (64 KB).
// grid(4, BB), block 256.
__global__ void __launch_bounds__(256)
k_upd(const __bf16* __restrict__ Fncur, const __bf16* __restrict__ AT,
      const float* __restrict__ F1f, __bf16* __restrict__ Fnnxt,
      __bf16* __restrict__ FnnxtT) {
  __shared__ __align__(16) __bf16 fnsh[RP * NN];  // 64 KB
  const int b = blockIdx.y;
  const int m0 = blockIdx.x * 128;
  const int w = threadIdx.x >> 5;
  const __bf16* fn = Fncur + (size_t)b * RP * NN;
  for (int idx = threadIdx.x; idx < RP * NN / 8; idx += 256)
    stage16(&fnsh[idx * 8], &fn[idx * 8]);
  stage_commit();

  const __bf16* at = AT + (size_t)b * NN * NN;
  const float* f1 = F1f + (size_t)b * RP * NN;
  __bf16* out = Fnnxt + (size_t)b * RP * NN;
  __bf16* outT = FnnxtT + (size_t)b * NN * RP;
  const int c0 = m0 + w * 16;
  v8f acc[4] = {};
  for (int k0 = 0; k0 < NN; k0 += 32) {
    if (k0 + 32 < NN) __builtin_prefetch(&at[(size_t)c0 * NN + k0 + 32], 0, 1);
    const v16bf bb = load_b16(k0,
        [&](int k, int c) { return at[(size_t)(c0 + c) * NN + k]; });
#pragma unroll
    for (int j = 0; j < 4; ++j) {
      const v16bf a = load_a16(k0,
          [&](int r, int k) { return fnsh[(j * 16 + r) * NN + k]; });
      acc[j] = wmma_mma(a, bb, acc[j]);
    }
  }
#pragma unroll
  for (int j = 0; j < 4; ++j)
    wmma_store(acc[j], [&](int r, int c, float v) {
      const int rr = j * 16 + r, cc = c0 + c;
      const __bf16 o = (__bf16)(v * f1[(size_t)rr * NN + cc]);
      out[(size_t)rr * NN + cc] = o;
      outT[(size_t)cc * RP + rr] = o;  // per-lane contiguous
    });
}

// fT[b,:] = fs[b,:]/max(||fs[b,:]||,1e-12) -> bf16. grid BB, block 256.
__global__ void __launch_bounds__(256)
k_norm(const float* __restrict__ fs, __bf16* __restrict__ fT) {
  __shared__ float red[256];
  const int b = blockIdx.x;
  const float* f = fs + (size_t)b * D0;
  const float v0 = f[threadIdx.x];
  const float v1 = f[threadIdx.x + 256];
  red[threadIdx.x] = v0 * v0 + v1 * v1;
  __syncthreads();
  for (int st = 128; st; st >>= 1) {
    if (threadIdx.x < st) red[threadIdx.x] += red[threadIdx.x + st];
    __syncthreads();
  }
  const float inv = fast_rcp(fmaxf(sqrtf(red[0]), 1e-12f));
  fT[(size_t)b * D0 + threadIdx.x] = (__bf16)(v0 * inv);
  fT[(size_t)b * D0 + threadIdx.x + 256] = (__bf16)(v1 * inv);
}

// out = act(in @ Wt^T + bias). Input (A operand) async-staged to LDS. grid N/128, block 256.
__global__ void __launch_bounds__(256)
k_mlp(const __bf16* __restrict__ inb, const __bf16* __restrict__ wb,
      const float* __restrict__ bias, int N, int K, int relu,
      __bf16* __restrict__ outb, float* __restrict__ outf) {
  __shared__ __align__(16) __bf16 insh[64 * D0];  // up to 64 KB
  const int jb = blockIdx.x * 128;
  const int w = threadIdx.x >> 5;
  for (int idx = threadIdx.x; idx < 64 * K / 8; idx += 256)
    stage16(&insh[idx * 8], &inb[idx * 8]);
  stage_commit();

  const int c0 = jb + w * 16;
  v8f acc[4] = {};
  for (int k0 = 0; k0 < K; k0 += 32) {
    const v16bf bb = load_b16(k0,
        [&](int k, int c) { return wb[(size_t)(c0 + c) * K + k]; });
#pragma unroll
    for (int j = 0; j < 4; ++j) {
      const v16bf a = load_a16(k0,
          [&](int r, int k) { return insh[(j * 16 + r) * K + k]; });
      acc[j] = wmma_mma(a, bb, acc[j]);
    }
  }
#pragma unroll
  for (int j = 0; j < 4; ++j)
    wmma_store(acc[j], [&](int r, int c, float v) {
      const int row = j * 16 + r, col = c0 + c;
      float x = v + bias[col];
      if (relu) x = fmaxf(x, 0.f);
      if (outb) outb[(size_t)row * N + col] = (__bf16)x;
      if (outf) outf[(size_t)row * N + col] = x;
    });
}

// ---------------- host launcher ----------------
extern "C" void kernel_launch(void* const* d_in, const int* in_sizes, int n_in,
                              void* d_out, int out_size, void* d_ws, size_t ws_size,
                              hipStream_t stream) {
  const float* node = (const float*)d_in[0];
  const int*   adj  = (const int*)d_in[1];
  const float* W    = (const float*)d_in[2];
  const float* Wv   = (const float*)d_in[3];
  const float* Ww   = (const float*)d_in[4];
  const float* Wg   = (const float*)d_in[5];
  const float* W0   = (const float*)d_in[6];
  const float* b0   = (const float*)d_in[7];
  const float* W1   = (const float*)d_in[8];
  const float* b1   = (const float*)d_in[9];
  const float* W2   = (const float*)d_in[10];
  const float* b2   = (const float*)d_in[11];
  const float* W3   = (const float*)d_in[12];
  const float* b3   = (const float*)d_in[13];

  char* p = (char*)d_ws;
  auto carve = [&](size_t bytes) {
    char* q = p;
    p += (bytes + 255) & ~(size_t)255;
    return q;
  };
  __bf16* wc    = (__bf16*)carve((size_t)RP * FEAT * 2);
  __bf16* wwb   = (__bf16*)carve((size_t)RP * RP * 2);
  __bf16* wgb   = (__bf16*)carve((size_t)RP * RP * 2);
  __bf16* w0b   = (__bf16*)carve((size_t)D0 * D0 * 2);
  __bf16* w1b   = (__bf16*)carve((size_t)D0 * D0 * 2);
  __bf16* w2b   = (__bf16*)carve((size_t)(D0 / 2) * D0 * 2);
  __bf16* w3b   = (__bf16*)carve((size_t)128 * (D0 / 2) * 2);
  float*  F1f   = (float*)carve((size_t)BB * RP * NN * 4);
  __bf16* Fn0   = (__bf16*)carve((size_t)BB * RP * NN * 2);
  __bf16* Fn0T  = (__bf16*)carve((size_t)BB * NN * RP * 2);
  __bf16* Fn1   = (__bf16*)carve((size_t)BB * RP * NN * 2);
  __bf16* Fn1T  = (__bf16*)carve((size_t)BB * NN * RP * 2);
  __bf16* WwFT  = (__bf16*)carve((size_t)BB * NN * RP * 2);
  float*  STf   = (float*)carve((size_t)BB * NN * NN * 4);
  __bf16* ATb   = (__bf16*)carve((size_t)BB * NN * NN * 2);
  float*  fs    = (float*)carve((size_t)BB * TT * RP * 4);
  __bf16* fTb   = (__bf16*)carve((size_t)BB * D0 * 2);
  __bf16* hAb   = (__bf16*)carve((size_t)BB * D0 * 2);
  __bf16* hBb   = (__bf16*)carve((size_t)BB * D0 * 2);

  // weight conversions
  k_cvt<<<(D0 * D0 + 255) / 256, 256, 0, stream>>>(W0, w0b, D0 * D0);
  k_cvt<<<(D0 * D0 + 255) / 256, 256, 0, stream>>>(W1, w1b, D0 * D0);
  k_cvt<<<((D0 / 2) * D0 + 255) / 256, 256, 0, stream>>>(W2, w2b, (D0 / 2) * D0);
  k_cvt<<<(128 * (D0 / 2) + 255) / 256, 256, 0, stream>>>(W3, w3b, 128 * (D0 / 2));
  k_prep<<<1, 256, 0, stream>>>(W, Wv, Ww, Wg, wc, wwb, wgb, fs);

  // F1 and initial Fn (both layouts)
  k_f1<<<dim3(8, BB), 128, 0, stream>>>(node, wc, F1f, Fn0, Fn0T);
  k_gate<<<dim3(4, BB), 256, 0, stream>>>(Fn0T, wgb, fs, 0);

  __bf16* fnbuf[2]  = {Fn0, Fn1};
  __bf16* fnTbuf[2] = {Fn0T, Fn1T};
  for (int it = 1; it < TT; ++it) {
    __bf16* cur  = fnbuf[(it - 1) & 1];
    __bf16* curT = fnTbuf[(it - 1) & 1];
    __bf16* nxt  = fnbuf[it & 1];
    __bf16* nxtT = fnTbuf[it & 1];
    k_wwf<<<dim3(4, BB), 256, 0, stream>>>(curT, wwb, WwFT);
    k_s<<<dim3(16, BB), 256, 0, stream>>>(curT, WwFT, adj, STf);
    k_softmax<<<dim3(64, BB), 256, 0, stream>>>(STf, ATb);
    k_upd<<<dim3(4, BB), 256, 0, stream>>>(cur, ATb, F1f, nxt, nxtT);
    k_gate<<<dim3(4, BB), 256, 0, stream>>>(nxtT, wgb, fs, it);
  }

  // normalize + MLP head
  k_norm<<<BB, 256, 0, stream>>>(fs, fTb);
  k_mlp<<<D0 / 128, 256, 0, stream>>>(fTb, w0b, b0, D0, D0, 1, hAb, nullptr);
  k_mlp<<<D0 / 128, 256, 0, stream>>>(hAb, w1b, b1, D0, D0, 1, hBb, nullptr);
  k_mlp<<<(D0 / 2) / 128, 256, 0, stream>>>(hBb, w2b, b2, D0 / 2, D0, 1, hAb, nullptr);
  k_mlp<<<1, 256, 0, stream>>>(hAb, w3b, b3, 128, D0 / 2, 0, nullptr, (float*)d_out);
}
